// WordLSTM_45483703665250
// MI455X (gfx1250) — matrix operand order repaired and checked by reading, not compiled, not run on hardware
//
#include <hip/hip_runtime.h>

// ---------- CDNA5 (gfx1250) wave32 WMMA types ----------
typedef __attribute__((ext_vector_type(16))) _Float16 v16h;
typedef __attribute__((ext_vector_type(8)))  _Float16 v8h;
typedef __attribute__((ext_vector_type(8)))  float    v8f;

#define DEV __device__ __forceinline__

// Problem constants (match reference)
constexpr int Bb = 64;      // batch
constexpr int Ss = 1024;    // seq len
constexpr int Ee = 256;     // embed
constexpr int Hh = 1024;    // hidden
constexpr int Gg = 4 * Hh;  // gates = 4096
constexpr int Pp = 256;     // projection size (== E)

// ---------- workspace layout (bytes), all chunks 256B aligned ----------
constexpr size_t SZ_XT   = (size_t)Ss * Bb * Ee * 2;   // x transposed (S,B,E) f16 : 33.5 MB
constexpr size_t SZ_WIH  = (size_t)2 * Gg * Ee * 2;    // 4 MB
constexpr size_t SZ_WHH  = (size_t)2 * Gg * Pp * 2;    // 4 MB
constexpr size_t SZ_WHR  = (size_t)2 * Pp * Hh * 2;    // 1 MB
constexpr size_t SZ_BIAS = (size_t)2 * Gg * 4;
constexpr size_t SZ_H    = (size_t)2 * Bb * Pp * 2;
constexpr size_t SZ_A    = (size_t)2 * Bb * Hh * 2;
constexpr size_t SZ_CNT  = 4096;                        // barrier counters
constexpr size_t OFF_XT   = 0;
constexpr size_t OFF_WIH  = OFF_XT + SZ_XT;
constexpr size_t OFF_WHH  = OFF_WIH + SZ_WIH;
constexpr size_t OFF_WHR  = OFF_WHH + SZ_WHH;
constexpr size_t OFF_BIAS = OFF_WHR + SZ_WHR;
constexpr size_t OFF_H    = OFF_BIAS + SZ_BIAS;
constexpr size_t OFF_A    = OFF_H + SZ_H;
constexpr size_t OFF_CNT  = OFF_A + SZ_A;

DEV float sigmoidf_(float x) { return 1.0f / (1.0f + __expf(-x)); }

// ---- fragment loaders matching CDNA5 WMMA VGPR layouts (CDNA5 ISA) ----
// A 16x32 f16: lanes 0-15: M=lane, K={kk..kk+7, kk+16..kk+23};
//              lanes 16-31: M=lane-16, K={kk+8..kk+15, kk+24..kk+31}.
DEV v16h load_a_frag(const _Float16* __restrict__ row, int kk, int hiHalf) {
    union { v16h v; v8h h[2]; } u;
    u.h[0] = *(const v8h*)(row + kk + hiHalf * 8);
    u.h[1] = *(const v8h*)(row + kk + 16 + hiHalf * 8);
    return u.v;
}
// B 32x16 f16: lanes 0-15: N=lane, K={kk..kk+7, kk+16..kk+23}; lanes 16-31: K={kk+8..kk+15, kk+24..kk+31}.
DEV v16h load_b_lds(const _Float16* row, int kk, int hiHalf) {
    union { v16h v; v8h h[2]; } u;
    const _Float16* p = row + kk + hiHalf * 8;
    u.h[0] = *(const v8h*)(p);
    u.h[1] = *(const v8h*)(p + 16);
    return u.v;
}

DEV v8f wmma16(v16h a, v16h b, v8f c) {
    return __builtin_amdgcn_wmma_f32_16x16x32_f16(false, a, false, b, (short)0, c, false, false);
}

// ---------------- setup kernels ----------------
__global__ void k_convert_x(const float* __restrict__ x, _Float16* __restrict__ xT) {
    int idx = blockIdx.x * 256 + threadIdx.x;        // over B*S*E
    int e  = idx & (Ee - 1);
    int bs = idx >> 8;
    int s  = bs & (Ss - 1);
    int b  = bs >> 10;
    xT[(((size_t)s * Bb + b) << 8) + e] = (_Float16)x[idx];
}

__global__ void k_f32_to_f16(const float* __restrict__ src, _Float16* __restrict__ dst, int n) {
    int i = blockIdx.x * 256 + threadIdx.x;
    if (i < n) dst[i] = (_Float16)src[i];
}

__global__ void k_init(const float* __restrict__ bihf, const float* __restrict__ bhhf,
                       const float* __restrict__ bihb, const float* __restrict__ bhhb,
                       float* __restrict__ bias, _Float16* __restrict__ h, int* __restrict__ cnt) {
    int i = blockIdx.x * 256 + threadIdx.x;          // grid covers 2*B*P = 32768
    if (i < 2 * Gg) {
        int g = i & (Gg - 1);
        bias[i] = (i < Gg) ? (bihf[g] + bhhf[g]) : (bihb[g] + bhhb[g]);
    }
    if (i < 2 * Bb * Pp) h[i] = (_Float16)0.0f;
    if (i < 1024) cnt[i] = 0;
}

// ---------------- persistent bidirectional scan kernel ----------------
// grid (68, 2): blockIdx.x < 64 -> gate/cell blocks; blockIdx.x >= 64 -> projection blocks.
// Cross-workgroup split barriers via monotonic atomic counters. Gate blocks compute the
// x-half of each step's GEMM BEFORE waiting on h (overlaps with previous projection).
__global__ void __launch_bounds__(128)
k_scan(const _Float16* __restrict__ xT,
       const _Float16* __restrict__ Wih, const _Float16* __restrict__ Whh,
       const _Float16* __restrict__ Whr, const float* __restrict__ bias,
       _Float16* hG, _Float16* aG, int* cnt, float* __restrict__ out)
{
    __shared__ __align__(16) _Float16 smem[64 * 1032];   // 132 KB (<= 320KB/WGP)

    const int dir    = blockIdx.y;
    const int tid    = threadIdx.x;
    const int lane   = tid & 31;
    const int m0     = (tid >> 5) << 4;
    const int n      = lane & 15;
    const int hiHalf = lane >> 4;

    int* g_done = cnt + dir * 256;        // gates blocks: +1 per step each (64/step)
    int* p_done = cnt + dir * 256 + 64;   // proj  blocks: +1 per step each (4/step)

    _Float16* hD = hG + (size_t)dir * (Bb * Pp);
    _Float16* aD = aG + (size_t)dir * (Bb * Hh);

    if (blockIdx.x < 64) {
        // =================== gates + cell update role ===================
        const int hu0 = blockIdx.x << 4;
        {   // LDS preload: row r=q*16+j holds [W_ih row g | W_hh row g], stride 520 (pad -> no bank conflicts)
            const _Float16* W0 = Wih + (size_t)dir * ((size_t)Gg * Ee);
            const _Float16* W1 = Whh + (size_t)dir * ((size_t)Gg * Pp);
            for (int u = tid; u < 64 * 64; u += 128) {
                int r = u >> 6, seg = u & 63;
                int g = (r >> 4) * Hh + hu0 + (r & 15);
                const _Float16* src = (seg < 32) ? (W0 + (size_t)g * 256 + seg * 8)
                                                 : (W1 + (size_t)g * 256 + (seg - 32) * 8);
                *(v8h*)(smem + r * 520 + seg * 8) = *(const v8h*)src;
            }
        }
        __syncthreads();

        const int    hu = hu0 + n;
        const size_t gb = (size_t)dir * Gg;
        const float  bi = bias[gb + 0 * Hh + hu];
        const float  bf = bias[gb + 1 * Hh + hu];
        const float  bg = bias[gb + 2 * Hh + hu];
        const float  bo = bias[gb + 3 * Hh + hu];

        float creg[8];
        #pragma unroll
        for (int r = 0; r < 8; ++r) creg[r] = 0.0f;

        const _Float16* hRow = hD + (size_t)(m0 + n) * Pp;
        const _Float16* lb0  = smem + (0 * 16 + n) * 520;
        const _Float16* lb1  = smem + (1 * 16 + n) * 520;
        const _Float16* lb2  = smem + (2 * 16 + n) * 520;
        const _Float16* lb3  = smem + (3 * 16 + n) * 520;

        for (int t = 0; t < Ss; ++t) {
            const int tx = dir ? (Ss - 1 - t) : t;
            const _Float16* xRow = xT + (size_t)tx * (Bb * Ee) + (size_t)(m0 + n) * Ee;
            if (t + 1 < Ss) {   // prefetch next step's A rows (global_prefetch_b8)
                const int tx2 = dir ? (Ss - 2 - t) : (t + 1);
                __builtin_prefetch(xT + (size_t)tx2 * (Bb * Ee) + (size_t)(m0 + n) * Ee, 0, 0);
            }

            // bias folded into accumulator init (C operand of first WMMA)
            v8f a0, a1, a2, a3;
            #pragma unroll
            for (int r = 0; r < 8; ++r) { a0[r] = bi; a1[r] = bf; a2[r] = bg; a3[r] = bo; }

            // ---- phase 1: x_t @ W_ih^T (independent of h -> overlaps prev projection) ----
            #pragma unroll
            for (int kk = 0; kk < 256; kk += 32) {
                v16h af = load_a_frag(xRow, kk, hiHalf);
                a0 = wmma16(af, load_b_lds(lb0, kk, hiHalf), a0);
                a1 = wmma16(af, load_b_lds(lb1, kk, hiHalf), a1);
                a2 = wmma16(af, load_b_lds(lb2, kk, hiHalf), a2);
                a3 = wmma16(af, load_b_lds(lb3, kk, hiHalf), a3);
            }

            // ---- wait for projection of step t-1 (h ready) ----
            if (t > 0) {
                if (tid == 0) {
                    while (__hip_atomic_load(p_done, __ATOMIC_ACQUIRE, __HIP_MEMORY_SCOPE_AGENT) < 4 * t)
                        __builtin_amdgcn_s_sleep(1);
                }
                __syncthreads();
                __threadfence();   // acquire for all threads: see fresh h
            }

            // ---- phase 2: h @ W_hh^T ----
            #pragma unroll
            for (int kk = 0; kk < 256; kk += 32) {
                v16h af = load_a_frag(hRow, kk, hiHalf);
                a0 = wmma16(af, load_b_lds(lb0 + 256, kk, hiHalf), a0);
                a1 = wmma16(af, load_b_lds(lb1 + 256, kk, hiHalf), a1);
                a2 = wmma16(af, load_b_lds(lb2 + 256, kk, hiHalf), a2);
                a3 = wmma16(af, load_b_lds(lb3 + 256, kk, hiHalf), a3);
            }

            // in-register LSTM cell update; c never leaves VGPRs
            #pragma unroll
            for (int r = 0; r < 8; ++r) {
                const int   m  = m0 + r + hiHalf * 8;
                const float iv = sigmoidf_(a0[r]);
                const float fv = sigmoidf_(a1[r]);
                const float gv = tanhf(a2[r]);
                const float ov = sigmoidf_(a3[r]);
                const float cv = fv * creg[r] + iv * gv;
                creg[r] = cv;
                aD[(size_t)m * Hh + hu] = (_Float16)(ov * tanhf(cv));
            }
            __threadfence();       // release our a-slice
            __syncthreads();
            if (tid == 0)
                __hip_atomic_fetch_add(g_done, 1, __ATOMIC_RELEASE, __HIP_MEMORY_SCOPE_AGENT);
        }
    } else {
        // =================== projection role: h = a @ W_hr^T ===================
        const int n0 = (blockIdx.x - 64) << 6;
        {   // LDS preload: 64 rows of W_hr (1024 f16 each), stride 1032 (pad)
            const _Float16* Wd = Whr + (size_t)dir * ((size_t)Pp * Hh);
            for (int u = tid; u < 64 * 128; u += 128) {
                int r = u >> 7, seg = u & 127;
                *(v8h*)(smem + r * 1032 + seg * 8) = *(const v8h*)(Wd + (size_t)(n0 + r) * Hh + seg * 8);
            }
        }
        __syncthreads();

        const _Float16* aRow = aD + (size_t)(m0 + n) * Hh;
        const _Float16* lb0  = smem + (0 * 16 + n) * 1032;
        const _Float16* lb1  = smem + (1 * 16 + n) * 1032;
        const _Float16* lb2  = smem + (2 * 16 + n) * 1032;
        const _Float16* lb3  = smem + (3 * 16 + n) * 1032;

        for (int t = 0; t < Ss; ++t) {
            if (tid == 0) {    // wait for all gate blocks of step t
                while (__hip_atomic_load(g_done, __ATOMIC_ACQUIRE, __HIP_MEMORY_SCOPE_AGENT) < 64 * (t + 1))
                    __builtin_amdgcn_s_sleep(1);
            }
            __syncthreads();
            __threadfence();   // acquire: see fresh a

            v8f a0 = {}, a1 = {}, a2 = {}, a3 = {};
            #pragma unroll 4
            for (int kk = 0; kk < Hh; kk += 32) {
                v16h af = load_a_frag(aRow, kk, hiHalf);
                a0 = wmma16(af, load_b_lds(lb0, kk, hiHalf), a0);
                a1 = wmma16(af, load_b_lds(lb1, kk, hiHalf), a1);
                a2 = wmma16(af, load_b_lds(lb2, kk, hiHalf), a2);
                a3 = wmma16(af, load_b_lds(lb3, kk, hiHalf), a3);
            }

            // critical path: only the f16 h-state must be visible before signaling
            #pragma unroll
            for (int r = 0; r < 8; ++r) {
                const int    m  = m0 + r + hiHalf * 8;
                const size_t hb = (size_t)m * Pp;
                hD[hb + n0 +  0 + n] = (_Float16)a0[r];
                hD[hb + n0 + 16 + n] = (_Float16)a1[r];
                hD[hb + n0 + 32 + n] = (_Float16)a2[r];
                hD[hb + n0 + 48 + n] = (_Float16)a3[r];
            }
            __threadfence();   // release h
            __syncthreads();
            if (tid == 0)
                __hip_atomic_fetch_add(p_done, 1, __ATOMIC_RELEASE, __HIP_MEMORY_SCOPE_AGENT);

            // off-critical-path: drain f32 outputs to HBM while gates proceed
            const int t_out = dir ? (Ss - 1 - t) : t;
            #pragma unroll
            for (int r = 0; r < 8; ++r) {
                const int    m  = m0 + r + hiHalf * 8;
                const size_t ob = ((size_t)m * Ss + t_out) * (2 * Pp) + (size_t)dir * Pp;
                out[ob + n0 +  0 + n] = a0[r];
                out[ob + n0 + 16 + n] = a1[r];
                out[ob + n0 + 32 + n] = a2[r];
                out[ob + n0 + 48 + n] = a3[r];
            }
        }
    }
}

// ---------------- host launch ----------------
extern "C" void kernel_launch(void* const* d_in, const int* in_sizes, int n_in,
                              void* d_out, int out_size, void* d_ws, size_t ws_size,
                              hipStream_t stream) {
    (void)in_sizes; (void)n_in; (void)out_size; (void)ws_size;
    const float* x     = (const float*)d_in[0];
    const float* Wih_f = (const float*)d_in[1];
    const float* Whh_f = (const float*)d_in[2];
    const float* bih_f = (const float*)d_in[3];
    const float* bhh_f = (const float*)d_in[4];
    const float* Whr_f = (const float*)d_in[5];
    const float* Wih_b = (const float*)d_in[6];
    const float* Whh_b = (const float*)d_in[7];
    const float* bih_b = (const float*)d_in[8];
    const float* bhh_b = (const float*)d_in[9];
    const float* Whr_b = (const float*)d_in[10];
    float* out = (float*)d_out;

    char* ws = (char*)d_ws;
    _Float16* xT   = (_Float16*)(ws + OFF_XT);
    _Float16* Wih  = (_Float16*)(ws + OFF_WIH);
    _Float16* Whh  = (_Float16*)(ws + OFF_WHH);
    _Float16* Whr  = (_Float16*)(ws + OFF_WHR);
    float*    bias = (float*)   (ws + OFF_BIAS);
    _Float16* hst  = (_Float16*)(ws + OFF_H);
    _Float16* ast  = (_Float16*)(ws + OFF_A);
    int*      cnt  = (int*)     (ws + OFF_CNT);

    // setup: f16 conversions, bias fold, state + barrier zeroing
    k_convert_x<<<(Bb * Ss * Ee) / 256, 256, 0, stream>>>(x, xT);
    k_f32_to_f16<<<(Gg * Ee + 255) / 256, 256, 0, stream>>>(Wih_f, Wih,                  Gg * Ee);
    k_f32_to_f16<<<(Gg * Ee + 255) / 256, 256, 0, stream>>>(Wih_b, Wih + (size_t)Gg * Ee, Gg * Ee);
    k_f32_to_f16<<<(Gg * Pp + 255) / 256, 256, 0, stream>>>(Whh_f, Whh,                  Gg * Pp);
    k_f32_to_f16<<<(Gg * Pp + 255) / 256, 256, 0, stream>>>(Whh_b, Whh + (size_t)Gg * Pp, Gg * Pp);
    k_f32_to_f16<<<(Pp * Hh + 255) / 256, 256, 0, stream>>>(Whr_f, Whr,                  Pp * Hh);
    k_f32_to_f16<<<(Pp * Hh + 255) / 256, 256, 0, stream>>>(Whr_b, Whr + (size_t)Pp * Hh, Pp * Hh);
    k_init<<<(2 * Bb * Pp) / 256, 256, 0, stream>>>(bih_f, bhh_f, bih_b, bhh_b, bias, hst, cnt);

    // one persistent kernel runs the full bidirectional scan (no per-step launches)
    k_scan<<<dim3(68, 2), 128, 0, stream>>>(xT, Wih, Whh, Whr, bias, hst, ast, cnt, out);
}
